// ResNetBKLayer_62319975465273
// MI455X (gfx1250) — compile-verified, matrix operand order checked by reading, not run on hardware
//
#include <hip/hip_runtime.h>
#include <hip/hip_bf16.h>

// ---------------------------------------------------------------------------
// Problem constants (from the reference): B=32, S=512, D=2048
// ---------------------------------------------------------------------------
#define BB   32
#define SS   512
#define DD   2048
#define NROW (BB * SS)        // 16384 flattened rows
#define MTILE 64              // x rows staged per block
#define ROWPAD (DD + 8)       // bf16 elements per LDS row (bank-conflict pad)

typedef __attribute__((ext_vector_type(16))) __bf16 v16bf;
typedef __attribute__((ext_vector_type(8)))  float  v8f;

union FragBF {                // 16 bf16 = 32 bytes = two 16B chunks
    v16bf v;
    uint4 q[2];
};

// ---------------------------------------------------------------------------
// Kernel 1: W1 (fp32, [K,N] row-major) -> W1T (bf16, [N,K] row-major)
// so that WMMA B-fragments are contiguous 16B loads along K.
// ---------------------------------------------------------------------------
__global__ __launch_bounds__(1024) void w1_transpose_kernel(
    const float* __restrict__ W1, __bf16* __restrict__ W1T)
{
    __shared__ float t[32][33];
    const int k = blockIdx.y * 32 + threadIdx.y;   // W1 row (K)
    const int n = blockIdx.x * 32 + threadIdx.x;   // W1 col (N)
    t[threadIdx.y][threadIdx.x] = W1[(size_t)k * DD + n];
    __syncthreads();
    const int ko = blockIdx.y * 32 + threadIdx.x;
    const int no = blockIdx.x * 32 + threadIdx.y;
    W1T[(size_t)no * DD + ko] = (__bf16)t[threadIdx.x][threadIdx.y];
}

// ---------------------------------------------------------------------------
// Kernel 2: fused  v = relu(x@W1 + b1) @ W2 + b2   (h never hits HBM)
// Block: 256 threads = 8 waves. Each wave owns a distinct 256-column N range
// (16 n-tiles of 16) and sweeps ALL 4 M-subtiles of the 64-row LDS x-tile,
// so one B fragment feeds 4 WMMAs and no two waves fetch the same W1T data.
// B (global) and A (LDS) fragments are both double-buffered so each WMMA only
// waits on loads issued a full WMMA earlier.
// ---------------------------------------------------------------------------
__global__ __launch_bounds__(256) void gemm_v_kernel(
    const float*  __restrict__ x,   const __bf16* __restrict__ W1T,
    const float*  __restrict__ b1,  const float*  __restrict__ W2,
    const float*  __restrict__ b2,  float* __restrict__ vout)
{
    extern __shared__ char smem[];
    __bf16* xs   = (__bf16*)smem;                                  // 64*2056 bf16
    float*  vbuf = (float*)(smem + (size_t)MTILE * ROWPAD * sizeof(__bf16)); // 8*64 f32

    const int tid  = threadIdx.x;
    const int row0 = blockIdx.x * MTILE;

    // ---- stage 64 x-rows (fp32) into LDS as bf16, padded rows -------------
    const float4* xg = (const float4*)(x + (size_t)row0 * DD);
    for (int i = tid; i < MTILE * (DD / 4); i += 256) {
        const int r = i >> 9;           // DD/4 == 512 float4 per row
        const int c = i & 511;
        const float4 f = xg[(size_t)r * (DD / 4) + c];
        __bf16* dst = xs + (size_t)r * ROWPAD + c * 4;
        dst[0] = (__bf16)f.x; dst[1] = (__bf16)f.y;
        dst[2] = (__bf16)f.z; dst[3] = (__bf16)f.w;
    }
    __syncthreads();

    const int wave = tid >> 5;
    const int lane = tid & 31;
    const int half = lane >> 4;    // lane half selects K sub-ranges (ISA layout)
    const int l16  = lane & 15;

    // 16-bit A 16x32 layout: per lane, VGPR0-3 = K half*8+0..7, VGPR4-7 = +16
    const __bf16* abase = xs + (size_t)l16 * ROWPAD + half * 8;
    const int MSTRIDE = 16 * ROWPAD;   // LDS elements between M-subtiles

    float vacc[4][8] = {};         // per-wave v partials: [msub][C-row r]

    const int nbase = wave * 256;  // this wave's private N range
    for (int nt = 0; nt < 16; ++nt) {
        const int n = nbase + nt * 16 + l16;             // this lane's column
        const __bf16* brow = W1T + (size_t)n * DD + half * 8;
        __builtin_prefetch(brow + (size_t)16 * DD, 0, 1); // next n-tile's row

        v8f acc0, acc1, acc2, acc3;
        {
            const v8f vz = {0.f, 0.f, 0.f, 0.f, 0.f, 0.f, 0.f, 0.f};
            acc0 = vz; acc1 = vz; acc2 = vz; acc3 = vz;
        }

        FragBF Bcur, Bnext, Aa, Ab;
        Bcur.q[0] = *(const uint4*)(brow);
        Bcur.q[1] = *(const uint4*)(brow + 16);
        Aa.q[0]   = *(const uint4*)(abase);              // m=0, k=0
        Aa.q[1]   = *(const uint4*)(abase + 16);

#pragma unroll 2
        for (int k = 0; k < DD; k += 32) {
            // next B fragment in flight under the 4 WMMAs (256B ws slack at tail)
            Bnext.q[0] = *(const uint4*)(brow + k + 32);
            Bnext.q[1] = *(const uint4*)(brow + k + 48);

            // m=1 A in flight, consume m=0
            Ab.q[0] = *(const uint4*)(abase + 1 * MSTRIDE + k);
            Ab.q[1] = *(const uint4*)(abase + 1 * MSTRIDE + k + 16);
            acc0 = __builtin_amdgcn_wmma_f32_16x16x32_bf16(
                false, Aa.v, false, Bcur.v, (short)0, acc0, false, false);

            // m=2 A in flight, consume m=1
            Aa.q[0] = *(const uint4*)(abase + 2 * MSTRIDE + k);
            Aa.q[1] = *(const uint4*)(abase + 2 * MSTRIDE + k + 16);
            acc1 = __builtin_amdgcn_wmma_f32_16x16x32_bf16(
                false, Ab.v, false, Bcur.v, (short)0, acc1, false, false);

            // m=3 A in flight, consume m=2
            Ab.q[0] = *(const uint4*)(abase + 3 * MSTRIDE + k);
            Ab.q[1] = *(const uint4*)(abase + 3 * MSTRIDE + k + 16);
            acc2 = __builtin_amdgcn_wmma_f32_16x16x32_bf16(
                false, Aa.v, false, Bcur.v, (short)0, acc2, false, false);

            // m=0 of next k in flight, consume m=3
            // (tail read at k=DD spans the row padding and the next LDS row; values unused)
            Aa.q[0] = *(const uint4*)(abase + k + 32);
            Aa.q[1] = *(const uint4*)(abase + k + 48);
            acc3 = __builtin_amdgcn_wmma_f32_16x16x32_bf16(
                false, Ab.v, false, Bcur.v, (short)0, acc3, false, false);

            Bcur = Bnext;
        }

        // C layout: VGPR r holds (M = msub*16 + r + 8*half, N = l16).
        const float b1n = b1[n];
        const float w2n = W2[n];
#pragma unroll
        for (int r = 0; r < 8; ++r) {
            float h0 = acc0[r] + b1n; h0 = h0 > 0.f ? h0 : 0.f;
            float h1 = acc1[r] + b1n; h1 = h1 > 0.f ? h1 : 0.f;
            float h2 = acc2[r] + b1n; h2 = h2 > 0.f ? h2 : 0.f;
            float h3 = acc3[r] + b1n; h3 = h3 > 0.f ? h3 : 0.f;
            vacc[0][r] += h0 * w2n;
            vacc[1][r] += h1 * w2n;
            vacc[2][r] += h2 * w2n;
            vacc[3][r] += h3 * w2n;
        }
    }

    // reduce over N (=16 lanes of each half) with wave32 shuffles
#pragma unroll
    for (int m = 0; m < 4; ++m)
#pragma unroll
        for (int r = 0; r < 8; ++r) {
            float s = vacc[m][r];
            s += __shfl_xor(s, 1, 32);
            s += __shfl_xor(s, 2, 32);
            s += __shfl_xor(s, 4, 32);
            s += __shfl_xor(s, 8, 32);
            vacc[m][r] = s;
        }
    if (l16 == 0) {
#pragma unroll
        for (int m = 0; m < 4; ++m)
#pragma unroll
            for (int r = 0; r < 8; ++r)
                vbuf[wave * MTILE + m * 16 + half * 8 + r] = vacc[m][r];
    }
    __syncthreads();
    if (tid < MTILE) {
        float s = b2[0];
#pragma unroll
        for (int w = 0; w < 8; ++w) s += vbuf[w * MTILE + tid];
        vout[row0 + tid] = s;
    }
}

// ---------------------------------------------------------------------------
// Kernel 3: Green's function diagonal via theta/phi continuants, complex fp64.
// One wave, lane b = batch b (32 batches). theta spilled to workspace.
// ---------------------------------------------------------------------------
struct cplx { double r, i; };
__device__ inline cplx cmul(cplx a, cplx b) {
    return {a.r * b.r - a.i * b.i, a.r * b.i + a.i * b.r};
}
__device__ inline cplx cdiv(cplx a, cplx b) {   // Smith's algorithm
    if (fabs(b.r) >= fabs(b.i)) {
        const double t = b.i / b.r, den = b.r + b.i * t;
        return {(a.r + a.i * t) / den, (a.i - a.r * t) / den};
    } else {
        const double t = b.r / b.i, den = b.r * t + b.i;
        return {(a.r * t + a.i) / den, (a.i * t - a.r) / den};
    }
}

__global__ void green_kernel(const float* __restrict__ v,
                             double2* __restrict__ thetaWS,
                             float* __restrict__ Gr, float* __restrict__ Gi)
{
    const int b = threadIdx.x;                 // 32 lanes = 32 batches
    const float* vb = v + b * SS;
    double2* th = thetaWS + (size_t)b * SS;

    // forward: theta[k] = a_{k-1}*theta[k-1] - theta[k-2], a_i = (v_i-2) - 1j
    cplx t0 = {1.0, 0.0};
    cplx t1 = {(double)vb[0] - 2.0, -1.0};
    th[0] = make_double2(t0.r, t0.i);
    th[1] = make_double2(t1.r, t1.i);
    for (int k = 2; k < SS; ++k) {
        const cplx a  = {(double)vb[k - 1] - 2.0, -1.0};
        const cplx t2 = {a.r * t1.r - a.i * t1.i - t0.r,
                         a.r * t1.i + a.i * t1.r - t0.i};
        th[k] = make_double2(t2.r, t2.i);
        t0 = t1; t1 = t2;
    }
    const cplx aN = {(double)vb[SS - 1] - 2.0, -1.0};
    cplx det = {aN.r * t1.r - aN.i * t1.i - t0.r,
                aN.r * t1.i + aN.i * t1.r - t0.i};
    det.r += 1e-18;

    // backward: phi[n-1]=1, phi[n-2]=a_{n-1}, phi[i]=a_{i+1}*phi[i+1]-phi[i+2]
    cplx p2 = {1.0, 0.0};
    cplx p1 = aN;
    {
        const double2 ti = th[SS - 1];
        const cplx g = cdiv(cmul({ti.x, ti.y}, p2), det);
        Gr[b * SS + SS - 1] = (float)g.r; Gi[b * SS + SS - 1] = (float)g.i;
    }
    {
        const double2 ti = th[SS - 2];
        const cplx g = cdiv(cmul({ti.x, ti.y}, p1), det);
        Gr[b * SS + SS - 2] = (float)g.r; Gi[b * SS + SS - 2] = (float)g.i;
    }
    for (int i = SS - 3; i >= 0; --i) {
        const cplx a  = {(double)vb[i + 1] - 2.0, -1.0};
        const cplx p0 = {a.r * p1.r - a.i * p1.i - p2.r,
                         a.r * p1.i + a.i * p1.r - p2.i};
        const double2 ti = th[i];
        const cplx g = cdiv(cmul({ti.x, ti.y}, p0), det);
        Gr[b * SS + i] = (float)g.r; Gi[b * SS + i] = (float)g.i;
        p2 = p1; p1 = p0;
    }
}

// ---------------------------------------------------------------------------
// Kernel 4: out[row, :] = gr*Wp[0,:] + gi*Wp[1,:] + bp   (streaming store)
// ---------------------------------------------------------------------------
__global__ __launch_bounds__(256) void out_kernel(
    const float* __restrict__ Gr, const float* __restrict__ Gi,
    const float* __restrict__ Wp, const float* __restrict__ bp,
    float* __restrict__ out)
{
    const int row = blockIdx.x;
    const float gr = Gr[row], gi = Gi[row];
    const float4* wp0 = (const float4*)Wp;
    const float4* wp1 = (const float4*)(Wp + DD);
    const float4* bpv = (const float4*)bp;
    float4* o = (float4*)(out + (size_t)row * DD);
    for (int c = threadIdx.x; c < DD / 4; c += 256) {
        const float4 a = wp0[c], b = wp1[c], d = bpv[c];
        float4 r;
        r.x = gr * a.x + gi * b.x + d.x;
        r.y = gr * a.y + gi * b.y + d.y;
        r.z = gr * a.z + gi * b.z + d.z;
        r.w = gr * a.w + gi * b.w + d.w;
        o[c] = r;
    }
}

// ---------------------------------------------------------------------------
// Launch
// ---------------------------------------------------------------------------
extern "C" void kernel_launch(void* const* d_in, const int* in_sizes, int n_in,
                              void* d_out, int out_size, void* d_ws, size_t ws_size,
                              hipStream_t stream)
{
    const float* x  = (const float*)d_in[0];
    const float* W1 = (const float*)d_in[1];
    const float* b1 = (const float*)d_in[2];
    const float* W2 = (const float*)d_in[3];
    const float* b2 = (const float*)d_in[4];
    const float* Wp = (const float*)d_in[5];
    const float* bp = (const float*)d_in[6];
    float* out = (float*)d_out;

    // workspace layout (~8.9 MB total)
    char* ws = (char*)d_ws;
    size_t off = 0;
    __bf16* W1T = (__bf16*)(ws + off);
    off += (size_t)DD * DD * sizeof(__bf16) + 256;      // +slack for B prefetch tail
    float*  v   = (float*)(ws + off);  off += (size_t)NROW * sizeof(float);
    off = (off + 255) & ~(size_t)255;
    double2* thetaWS = (double2*)(ws + off); off += (size_t)NROW * sizeof(double2);
    float* Gr = (float*)(ws + off); off += (size_t)NROW * sizeof(float);
    float* Gi = (float*)(ws + off); off += (size_t)NROW * sizeof(float);

    // 1) W1 -> bf16 transposed
    w1_transpose_kernel<<<dim3(DD / 32, DD / 32), dim3(32, 32), 0, stream>>>(W1, W1T);

    // 2) fused bf16-WMMA GEMM + bias + ReLU + dot(W2) -> v   (~265 KB LDS)
    const size_t smem = (size_t)MTILE * ROWPAD * sizeof(__bf16)
                      + 8 * MTILE * sizeof(float);
    (void)hipFuncSetAttribute(reinterpret_cast<const void*>(gemm_v_kernel),
                              hipFuncAttributeMaxDynamicSharedMemorySize, (int)smem);
    gemm_v_kernel<<<NROW / MTILE, 256, smem, stream>>>(x, W1T, b1, W2, b2, v);

    // 3) complex-double continuant scans (one wave, lane = batch)
    green_kernel<<<1, 32, 0, stream>>>(v, thetaWS, Gr, Gi);

    // 4) feats @ Wp + bp, streaming
    out_kernel<<<NROW, 256, 0, stream>>>(Gr, Gi, Wp, bp, out);
}
